// ShiftedWindowMSA_33011118637381
// MI455X (gfx1250) — matrix operand order repaired and checked by reading, not compiled, not run on hardware
//
#include <hip/hip_runtime.h>
#include <hip/hip_bf16.h>

// ---------------------------------------------------------------------------
// Shifted-window MSA for MI455X (gfx1250): all GEMM work on
// v_wmma_f32_16x16x32_bf16 (fp32 accumulate), wave32, 4 waves/block.
// Staging is register-pipelined: next K-tile global loads overlap WMMAs.
// ---------------------------------------------------------------------------

typedef __bf16 bf16_t;
typedef __attribute__((ext_vector_type(2)))  float  v2f;
typedef __attribute__((ext_vector_type(2)))  __bf16 v2bf;
typedef __attribute__((ext_vector_type(8)))  __bf16 v8bf;
typedef __attribute__((ext_vector_type(16))) __bf16 v16bf;
typedef __attribute__((ext_vector_type(8)))  float  v8f;

__device__ __forceinline__ bf16_t f2bf(float f) { return (bf16_t)f; }

// Packed f32x2 -> bf16x2: vector convert selects a single v_cvt_pk_bf16_f32.
__device__ __forceinline__ unsigned f2bf2(float a, float b) {
  v2f in; in[0] = a; in[1] = b;
  v2bf out = __builtin_convertvector(in, v2bf);
  return __builtin_bit_cast(unsigned, out);
}

// Build a 16-bit A-fragment from two contiguous 8-elem chunks (contiguous
// layout: lane half h covers K = h*16..h*16+15).
__device__ __forceinline__ v16bf make_afrag(const bf16_t* row, int h) {
  v8bf lo = *(const v8bf*)(row + h * 16);
  v8bf hi = *(const v8bf*)(row + h * 16 + 8);
  v16bf a;
#pragma unroll
  for (int e = 0; e < 8; ++e) { a[e] = lo[e]; a[e + 8] = hi[e]; }
  return a;
}

// ---------------------------------------------------------------------------
// Kernel 1: y = x @ W1^T + b1 (M=65536, N=768, K=256), bf16 WMMA.
// Epilogue: roll(-4,-4), window partition, head split, scatter into
// Q (x 1/sqrt(32)), K, and V^T buffers (bf16).
// ---------------------------------------------------------------------------
__global__ __launch_bounds__(128) void qkv_gemm_kernel(
    const float* __restrict__ x, const float* __restrict__ W1,
    const float* __restrict__ b1,
    bf16_t* __restrict__ qbuf, bf16_t* __restrict__ kbuf,
    bf16_t* __restrict__ vbuf) {
  __shared__ __align__(32) bf16_t As[64][32];
  __shared__ __align__(32) bf16_t Bs[64][32];

  const int tid  = threadIdx.x;
  const int lane = tid & 31;
  const int wave = tid >> 5;
  const int wm = wave >> 1, wn = wave & 1;
  const int h  = lane >> 4, ln = lane & 15;
  const int m0 = blockIdx.y * 64;
  const int n0 = blockIdx.x * 64;

  v8f acc[2][2] = {};
  float4 xa[4], wa[4];

  // issue all 8 tile loads back-to-back (no per-load waits)
  auto load_tile = [&](int k0) {
#pragma unroll
    for (int it = 0; it < 4; ++it) {
      int i4 = it * 128 + tid;            // float4 index 0..511
      int r = i4 >> 3, c = (i4 & 7) << 2;
      xa[it] = *(const float4*)(x  + (size_t)(m0 + r) * 256 + k0 + c);
      wa[it] = *(const float4*)(W1 + (size_t)(n0 + r) * 256 + k0 + c);
    }
  };
  auto store_tile = [&]() {
#pragma unroll
    for (int it = 0; it < 4; ++it) {
      int i4 = it * 128 + tid;
      int r = i4 >> 3, c = (i4 & 7) << 2;
      uint2 px; px.x = f2bf2(xa[it].x, xa[it].y); px.y = f2bf2(xa[it].z, xa[it].w);
      *(uint2*)&As[r][c] = px;
      uint2 pw; pw.x = f2bf2(wa[it].x, wa[it].y); pw.y = f2bf2(wa[it].z, wa[it].w);
      *(uint2*)&Bs[r][c] = pw;
    }
  };

  load_tile(0);
  for (int k0 = 0; k0 < 256; k0 += 32) {
    store_tile();
    __syncthreads();
    if (k0 + 32 < 256) load_tile(k0 + 32);   // overlap next tile with WMMAs

    v16bf af[2], bf[2];
#pragma unroll
    for (int t = 0; t < 2; ++t) {
      af[t] = make_afrag(&As[wm * 32 + t * 16 + ln][0], h);
      bf[t] = *(const v16bf*)&Bs[wn * 32 + t * 16 + ln][h * 16];
    }
#pragma unroll
    for (int ti = 0; ti < 2; ++ti)
#pragma unroll
      for (int tj = 0; tj < 2; ++tj)
        acc[ti][tj] = __builtin_amdgcn_wmma_f32_16x16x32_bf16(
            false, af[ti], false, bf[tj], (short)0, acc[ti][tj], false, false);
    __syncthreads();
  }

  const float qscale = 0.17677669529663687f;  // 1/sqrt(32)
#pragma unroll
  for (int ti = 0; ti < 2; ++ti)
#pragma unroll
    for (int tj = 0; tj < 2; ++tj)
#pragma unroll
      for (int v = 0; v < 8; ++v) {
        int mrow = m0 + wm * 32 + ti * 16 + v + 8 * h;
        int cg   = n0 + wn * 32 + tj * 16 + ln;
        float val = acc[ti][tj][v] + b1[cg];
        int qkv = cg % 3, eidx = cg / 3;
        int hh = eidx & 7, es = eidx >> 3;          // E -> (e=32 outer, H=8 inner)
        int b = mrow >> 12, s = mrow & 4095;
        int i0 = s >> 6, j0 = s & 63;
        int i = (i0 + 60) & 63, j = (j0 + 60) & 63;  // roll(-4,-4)
        int wi = i >> 3, r = i & 7, wj = j >> 3, cc = j & 7;
        int q = r * 8 + cc;
        size_t win = (((size_t)b * 8 + hh) * 8 + wi) * 8 + wj;
        if (qkv == 0)      qbuf[(win * 64 + q) * 32 + es] = f2bf(val * qscale);
        else if (qkv == 1) kbuf[(win * 64 + q) * 32 + es] = f2bf(val);
        else               vbuf[(win * 32 + es) * 64 + q] = f2bf(val);  // V^T
      }
}

// ---------------------------------------------------------------------------
// Kernel 2: per-(b,h,window) attention. 4 waves = 4 query strips of 16.
// scores WMMA accumulator pre-seeded with rel-pos bias + shift masks.
// ---------------------------------------------------------------------------
__global__ __launch_bounds__(128) void swin_attn_kernel(
    const bf16_t* __restrict__ qbuf, const bf16_t* __restrict__ kbuf,
    const bf16_t* __restrict__ vbuf, const float* __restrict__ pos_emb,
    bf16_t* __restrict__ attn) {
  __shared__ __align__(32) bf16_t probsL[4][16][64];
  __shared__ float pe[225];

  const int tid  = threadIdx.x;
  const int lane = tid & 31;
  const int wave = tid >> 5;
  const int h = lane >> 4, ln = lane & 15;

  const int blk = blockIdx.x;            // 0..8191
  const int wj = blk & 7, wi = (blk >> 3) & 7;
  const int hh = (blk >> 6) & 7, b = blk >> 9;

  for (int t = tid; t < 225; t += 128) pe[t] = pos_emb[t];
  __syncthreads();

  const size_t win = (((size_t)b * 8 + hh) * 8 + wi) * 8 + wj;
  const bf16_t* Qw = qbuf + win * 2048;   // [token][e]
  const bf16_t* Kw = kbuf + win * 2048;   // [token][e]
  const bf16_t* Vw = vbuf + win * 2048;   // [e][token]

  const int qi = wave;
  v16bf qf = make_afrag(Qw + (qi * 16 + ln) * 32, h);

  // ---- scores = Q K^T * (1/sqrt e) + bias + masks, via C-preload ----
  v8f sc[4];
#pragma unroll
  for (int kj = 0; kj < 4; ++kj) {
#pragma unroll
    for (int v = 0; v < 8; ++v) {
      int q = qi * 16 + v + 8 * h;
      int k = kj * 16 + ln;
      int qr = q >> 3, qc = q & 7, kr = k >> 3, kc = k & 7;
      float bias = pe[(kr - qr + 7) * 15 + (kc - qc + 7)];
      if (wi == 7 && ((qr >= 4) != (kr >= 4))) bias = -1e30f;
      if (wj == 7 && ((qc >= 4) != (kc >= 4))) bias = -1e30f;
      sc[kj][v] = bias;
    }
    v16bf kf = *(const v16bf*)(Kw + (kj * 16 + ln) * 32 + h * 16);
    sc[kj] = __builtin_amdgcn_wmma_f32_16x16x32_bf16(
        false, qf, false, kf, (short)0, sc[kj], false, false);
  }

  // ---- row softmax: reduce over 4 local cols x 16 lanes (half-wave) ----
#pragma unroll
  for (int v = 0; v < 8; ++v) {
    float m = fmaxf(fmaxf(sc[0][v], sc[1][v]), fmaxf(sc[2][v], sc[3][v]));
#pragma unroll
    for (int off = 1; off < 16; off <<= 1) m = fmaxf(m, __shfl_xor(m, off, 32));
    float p[4], s = 0.f;
#pragma unroll
    for (int kj = 0; kj < 4; ++kj) { p[kj] = __expf(sc[kj][v] - m); s += p[kj]; }
#pragma unroll
    for (int off = 1; off < 16; off <<= 1) s += __shfl_xor(s, off, 32);
    float rinv = 1.0f / s;
    int row = v + 8 * h;
#pragma unroll
    for (int kj = 0; kj < 4; ++kj)
      probsL[wave][row][kj * 16 + ln] = f2bf(p[kj] * rinv);
  }

  // ---- out = P V  (K=64 -> 2 k-chunks; N=32 -> 2 col tiles) ----
  v8f o[2] = {};
#pragma unroll
  for (int kc = 0; kc < 2; ++kc) {
    v16bf pf = make_afrag(&probsL[wave][ln][kc * 32], h);
#pragma unroll
    for (int nj = 0; nj < 2; ++nj) {
      v16bf vf = *(const v16bf*)(Vw + (nj * 16 + ln) * 64 + kc * 32 + h * 16);
      o[nj] = __builtin_amdgcn_wmma_f32_16x16x32_bf16(
          false, pf, false, vf, (short)0, o[nj], false, false);
    }
  }

  // ---- write to (B,hh,ww,H*e) layout, bf16, for final projection ----
#pragma unroll
  for (int nj = 0; nj < 2; ++nj)
#pragma unroll
    for (int v = 0; v < 8; ++v) {
      int q = qi * 16 + v + 8 * h;
      int r = q >> 3, c = q & 7;
      int e = nj * 16 + ln;
      size_t rowg = ((size_t)b * 64 + wi * 8 + r) * 64 + (wj * 8 + c);
      attn[rowg * 256 + hh * 32 + e] = f2bf(o[nj][v]);
    }
}

// ---------------------------------------------------------------------------
// Kernel 3: out = attn @ W2^T + b2 (M=65536, N=256, K=256), fp32 output.
// ---------------------------------------------------------------------------
__global__ __launch_bounds__(128) void proj_gemm_kernel(
    const bf16_t* __restrict__ attn, const float* __restrict__ W2,
    const float* __restrict__ b2, float* __restrict__ out) {
  __shared__ __align__(32) bf16_t As[64][32];
  __shared__ __align__(32) bf16_t Bs[64][32];

  const int tid  = threadIdx.x;
  const int lane = tid & 31;
  const int wave = tid >> 5;
  const int wm = wave >> 1, wn = wave & 1;
  const int h = lane >> 4, ln = lane & 15;
  const int m0 = blockIdx.y * 64;
  const int n0 = blockIdx.x * 64;

  v8f acc[2][2] = {};
  v8bf aa[2];
  float4 wa[4];

  auto load_tile = [&](int k0) {
#pragma unroll
    for (int it = 0; it < 2; ++it) {      // A: already bf16, 16B chunks
      int i8 = it * 128 + tid;            // v8bf index 0..255
      int r = i8 >> 2, c = (i8 & 3) << 3;
      aa[it] = *(const v8bf*)(attn + (size_t)(m0 + r) * 256 + k0 + c);
    }
#pragma unroll
    for (int it = 0; it < 4; ++it) {      // B: W2 f32
      int i4 = it * 128 + tid;
      int r = i4 >> 3, c = (i4 & 7) << 2;
      wa[it] = *(const float4*)(W2 + (size_t)(n0 + r) * 256 + k0 + c);
    }
  };
  auto store_tile = [&]() {
#pragma unroll
    for (int it = 0; it < 2; ++it) {
      int i8 = it * 128 + tid;
      int r = i8 >> 2, c = (i8 & 3) << 3;
      *(v8bf*)&As[r][c] = aa[it];
    }
#pragma unroll
    for (int it = 0; it < 4; ++it) {
      int i4 = it * 128 + tid;
      int r = i4 >> 3, c = (i4 & 7) << 2;
      uint2 pw; pw.x = f2bf2(wa[it].x, wa[it].y); pw.y = f2bf2(wa[it].z, wa[it].w);
      *(uint2*)&Bs[r][c] = pw;
    }
  };

  load_tile(0);
  for (int k0 = 0; k0 < 256; k0 += 32) {
    store_tile();
    __syncthreads();
    if (k0 + 32 < 256) load_tile(k0 + 32);

    v16bf af[2], bf[2];
#pragma unroll
    for (int t = 0; t < 2; ++t) {
      af[t] = make_afrag(&As[wm * 32 + t * 16 + ln][0], h);
      bf[t] = *(const v16bf*)&Bs[wn * 32 + t * 16 + ln][h * 16];
    }
#pragma unroll
    for (int ti = 0; ti < 2; ++ti)
#pragma unroll
      for (int tj = 0; tj < 2; ++tj)
        acc[ti][tj] = __builtin_amdgcn_wmma_f32_16x16x32_bf16(
            false, af[ti], false, bf[tj], (short)0, acc[ti][tj], false, false);
    __syncthreads();
  }

#pragma unroll
  for (int ti = 0; ti < 2; ++ti)
#pragma unroll
    for (int tj = 0; tj < 2; ++tj)
#pragma unroll
      for (int v = 0; v < 8; ++v) {
        int mrow = m0 + wm * 32 + ti * 16 + v + 8 * h;
        int cg   = n0 + wn * 32 + tj * 16 + ln;
        out[(size_t)mrow * 256 + cg] = acc[ti][tj][v] + b2[cg];
      }
}

// ---------------------------------------------------------------------------
extern "C" void kernel_launch(void* const* d_in, const int* in_sizes, int n_in,
                              void* d_out, int out_size, void* d_ws, size_t ws_size,
                              hipStream_t stream) {
  const float* x       = (const float*)d_in[0];
  const float* W1      = (const float*)d_in[1];
  const float* b1      = (const float*)d_in[2];
  const float* W2      = (const float*)d_in[3];
  const float* b2      = (const float*)d_in[4];
  const float* pos_emb = (const float*)d_in[5];

  // Workspace: Q,K,V^T,attn in bf16: 4 x 16,777,216 elems = 128 MB.
  const size_t NELEM = (size_t)16 * 8 * 64 * 64 * 32;  // B*H*Wh*Wh*n*e
  bf16_t* qbuf = (bf16_t*)d_ws;
  bf16_t* kbuf = qbuf + NELEM;
  bf16_t* vbuf = kbuf + NELEM;
  bf16_t* attn = vbuf + NELEM;

  qkv_gemm_kernel<<<dim3(12, 1024), 128, 0, stream>>>(x, W1, b1, qbuf, kbuf, vbuf);
  swin_attn_kernel<<<dim3(8192), 128, 0, stream>>>(qbuf, kbuf, vbuf, pos_emb, attn);
  proj_gemm_kernel<<<dim3(4, 1024), 128, 0, stream>>>(attn, W2, b2, (float*)d_out);
}
